// DynPredNet_84069689852208
// MI455X (gfx1250) — matrix operand order repaired and checked by reading, not compiled, not run on hardware
//
#include <hip/hip_runtime.h>

// ---------------- problem dims ----------------
#define NB    128
#define NT    6
#define IND   1024
#define RD    256
#define R2D   128
#define MIXD  16
#define HIDD  256
#define NITER 20

typedef __attribute__((ext_vector_type(16))) __bf16 v16bf;
typedef __attribute__((ext_vector_type(8)))  __bf16 v8bf;
typedef __attribute__((ext_vector_type(8)))  float  v8f;
typedef __attribute__((ext_vector_type(4)))  unsigned u32x4;
typedef unsigned short u16;

// acc scalar block: 0:|dr|^2 1:|r|^2 2:|dr2|^2 3:|r2|^2 4:conv 5:t 6:mf_sum
//                   7:temp_loss 8:temp2_loss

__device__ __forceinline__ u16 f2bfbits(float f) {
  unsigned u = __builtin_bit_cast(unsigned, f);
  unsigned r = u + 0x7FFFu + ((u >> 16) & 1u);   // round-to-nearest-even
  return (u16)(r >> 16);
}

// A tile 16x32 (MxK) from bf16 row-major [M,K]; ISA A layout:
// lane<16: K runs [k0..k0+7] and [k0+16..k0+23]; lane>=16: +8. Two b128 loads.
__device__ __forceinline__ v16bf load_abf(const u16* M, int ld, int r0, int c0) {
  int lane = threadIdx.x & 31;
  const u16* p = M + (size_t)(r0 + (lane & 15)) * ld + c0 + ((lane >> 4) << 3);
  v8bf lo = __builtin_bit_cast(v8bf, *(const u32x4*)p);
  v8bf hi = __builtin_bit_cast(v8bf, *(const u32x4*)(p + 16));
  return __builtin_shufflevector(lo, hi, 0, 1, 2, 3, 4, 5, 6, 7,
                                 8, 9, 10, 11, 12, 13, 14, 15);
}

// B operand: Bt = B^T [N,K] row-major; lane<16 holds K [k0..k0+7] and [k0+16..k0+23],
// lane>=16: +8 (same as A). Two b128 loads.
__device__ __forceinline__ v16bf load_bbf(const u16* Bt, int ld, int n0, int k0) {
  int lane = threadIdx.x & 31;
  const u16* p = Bt + (size_t)(n0 + (lane & 15)) * ld + k0 + ((lane >> 4) << 3);
  v8bf lo = __builtin_bit_cast(v8bf, *(const u32x4*)p);
  v8bf hi = __builtin_bit_cast(v8bf, *(const u32x4*)(p + 16));
  return __builtin_shufflevector(lo, hi, 0, 1, 2, 3, 4, 5, 6, 7,
                                 8, 9, 10, 11, 12, 13, 14, 15);
}

// 16x64 output per wave: issue A + all 4 B loads first, then 4 WMMAs, so the
// backend can use partial load waits and overlap loads with in-flight WMMAs.
template <int K>
__device__ __forceinline__ void mm4(const u16* A, int lda, const u16* Bt, int ldb,
                                    int r0, int c0, v8f acc[4]) {
#pragma unroll 4
  for (int k = 0; k < K; k += 32) {
    v16bf a = load_abf(A, lda, r0, k);
    v16bf b[4];
#pragma unroll
    for (int j = 0; j < 4; ++j) b[j] = load_bbf(Bt, ldb, c0 + 16 * j, k);
#pragma unroll
    for (int j = 0; j < 4; ++j)
      acc[j] = __builtin_amdgcn_wmma_f32_16x16x32_bf16(false, a, false, b[j], (short)0,
                                                       acc[j], false, false);
  }
}

template <int K>
__device__ __forceinline__ v8f mm1(const u16* A, int lda, const u16* Bt, int ldb,
                                   int r0, int c0) {
  v8f acc = {0.f, 0.f, 0.f, 0.f, 0.f, 0.f, 0.f, 0.f};
#pragma unroll 4
  for (int k = 0; k < K; k += 32) {
    v16bf a = load_abf(A, lda, r0, k);
    v16bf b = load_bbf(Bt, ldb, c0, k);
    acc = __builtin_amdgcn_wmma_f32_16x16x32_bf16(false, a, false, b, (short)0, acc,
                                                  false, false);
  }
  return acc;
}

__device__ __forceinline__ float blk_sum(float v, float* s) {
  int t = threadIdx.x;
  s[t] = v; __syncthreads();
  for (int k = 128; k > 0; k >>= 1) { if (t < k) s[t] += s[t + k]; __syncthreads(); }
  float r = s[0]; __syncthreads();
  return r;
}

// ---------------- utility kernels ----------------
__global__ void kz(float* p, int n) {
  int i = blockIdx.x * 256 + threadIdx.x;
  if (i < n) p[i] = 0.f;
}
__global__ void kz16(u16* p, int n) {
  int i = blockIdx.x * 256 + threadIdx.x;
  if (i < n) p[i] = 0;
}
__global__ void kcopy(float* d, const float* s, int n) {
  int i = blockIdx.x * 256 + threadIdx.x;
  if (i < n) d[i] = s[i];
}
__global__ void kcvt(const float* s, u16* d, int n) {
  int i = blockIdx.x * 256 + threadIdx.x;
  if (i < n) d[i] = f2bfbits(s[i]);
}
__global__ void kcvtT(const float* s, u16* d, int rows, int cols) {
  int i = blockIdx.x * 256 + threadIdx.x;
  if (i < rows * cols) {
    int r = i / cols, c = i % cols;
    d[(size_t)c * rows + r] = f2bfbits(s[i]);
  }
}

// ---------------- WMMA GEMM kernels (16x64 per wave) ----------------
// e = x - r @ W_sd  -> bf16 only (e is consumed solely as a GEMM A-operand)
__global__ void k_err(const float* xb, int ldx, const u16* rbf, const u16* WsdT,
                      u16* ebf) {
  int wave = blockIdx.x * 8 + (threadIdx.x >> 5);  // 128 waves, grid 16
  int tm = wave >> 4, tg = wave & 15;
  v8f z = {0.f, 0.f, 0.f, 0.f, 0.f, 0.f, 0.f, 0.f};
  v8f acc[4] = {z, z, z, z};
  mm4<RD>(rbf, RD, WsdT, RD, tm * 16, tg * 64, acc);
  int lane = threadIdx.x & 31;
  int row0 = tm * 16 + ((lane >> 4) << 3);
#pragma unroll
  for (int j = 0; j < 4; ++j) {
    int col = tg * 64 + j * 16 + (lane & 15);
#pragma unroll
    for (int v = 0; v < 8; ++v) {
      int row = row0 + v;
      ebf[row * IND + col] = f2bfbits(xb[(size_t)row * ldx + col] - acc[j][v]);
    }
  }
}

// g = e @ W_sd^T, fused ISTA update of r (+ bf16 shadow) + conv norms
__global__ void k_update_r(const u16* ebf, const u16* WsdB, float* r, u16* rbf,
                           const float* pre, float* accs, int use_temp) {
  if (accs[4] != 0.f) return;  // frozen (wave-uniform, EXEC stays full)
  int wave = blockIdx.x * 8 + (threadIdx.x >> 5);  // 32 waves, grid 4
  int tm = wave >> 2, tg = wave & 3;
  v8f z = {0.f, 0.f, 0.f, 0.f, 0.f, 0.f, 0.f, 0.f};
  v8f acc[4] = {z, z, z, z};
  mm4<IND>(ebf, IND, WsdB, IND, tm * 16, tg * 64, acc);
  int lane = threadIdx.x & 31;
  int row0 = tm * 16 + ((lane >> 4) << 3);
  float dsq = 0.f, nsq = 0.f;
#pragma unroll
  for (int j = 0; j < 4; ++j) {
    int col = tg * 64 + j * 16 + (lane & 15);
#pragma unroll
    for (int v = 0; v < 8; ++v) {
      int idx = (row0 + v) * RD + col;
      float rv = r[idx];
      float grad = (-2.0f / 128.0f) * acc[j][v];
      if (use_temp) grad += (2.0f / 128.0f) * (rv - fmaxf(pre[idx], 0.f));  // TW=1
      float zz = rv - 1e-2f * grad;
      float rn = fmaxf(fmaxf(zz - 1e-3f, 0.f) - fmaxf(-zz - 1e-3f, 0.f), 0.f);
      dsq += (rn - rv) * (rn - rv);
      nsq += rv * rv;
      r[idx] = rn;
      rbf[idx] = f2bfbits(rn);
    }
  }
  atomicAdd(&accs[0], dsq);
  atomicAdd(&accs[1], nsq);
}

// D(f32) = A(bf16) @ B (+bias); NG = Ntiles/4 column groups
template <int K, int NG>
__global__ void k_gemm4(const u16* A, int lda, const u16* Bt, int ldb,
                        const float* bias, float* D, int ldd) {
  int wave = blockIdx.x * 8 + (threadIdx.x >> 5);
  int tm = wave / NG, tg = wave % NG;
  v8f z = {0.f, 0.f, 0.f, 0.f, 0.f, 0.f, 0.f, 0.f};
  v8f acc[4] = {z, z, z, z};
  mm4<K>(A, lda, Bt, ldb, tm * 16, tg * 64, acc);
  int lane = threadIdx.x & 31;
  int row0 = tm * 16 + ((lane >> 4) << 3);
#pragma unroll
  for (int j = 0; j < 4; ++j) {
    int col = tg * 64 + j * 16 + (lane & 15);
    float bv = bias ? bias[col] : 0.f;
#pragma unroll
    for (int v = 0; v < 8; ++v) D[(row0 + v) * ldd + col] = acc[j][v] + bv;
  }
}

// bf16-output variant (for activations consumed only as GEMM A-operands)
template <int K, int NG>
__global__ void k_gemm4h(const u16* A, int lda, const u16* Bt, int ldb,
                         const float* bias, u16* D, int ldd) {
  int wave = blockIdx.x * 8 + (threadIdx.x >> 5);
  int tm = wave / NG, tg = wave % NG;
  v8f z = {0.f, 0.f, 0.f, 0.f, 0.f, 0.f, 0.f, 0.f};
  v8f acc[4] = {z, z, z, z};
  mm4<K>(A, lda, Bt, ldb, tm * 16, tg * 64, acc);
  int lane = threadIdx.x & 31;
  int row0 = tm * 16 + ((lane >> 4) << 3);
#pragma unroll
  for (int j = 0; j < 4; ++j) {
    int col = tg * 64 + j * 16 + (lane & 15);
    float bv = bias ? bias[col] : 0.f;
#pragma unroll
    for (int v = 0; v < 8; ++v) D[(row0 + v) * ldd + col] = f2bfbits(acc[j][v] + bv);
  }
}

// w = a2 @ hW3 + hb3 : N=16 single column tile, 8 waves
__global__ void k_h3(const u16* A, const u16* Bt, const float* bias, float* D) {
  int wave = threadIdx.x >> 5;
  v8f acc = mm1<HIDD>(A, HIDD, Bt, HIDD, wave * 16, 0);
  int lane = threadIdx.x & 31;
  int col = lane & 15;
  int row0 = wave * 16 + ((lane >> 4) << 3);
  float bv = bias[col];
#pragma unroll
  for (int v = 0; v < 8; ++v) D[(row0 + v) * MIXD + col] = acc[v] + bv;
}

// tj[b,m,:] = r_p @ temporal_m^T
__global__ void k_tj(const u16* rpbf, const u16* tempb, float* tj) {
  int m = blockIdx.y;
  int wave = blockIdx.x * 8 + (threadIdx.x >> 5);  // 32 waves, grid (4,16)
  int tm = wave >> 2, tg = wave & 3;
  v8f z = {0.f, 0.f, 0.f, 0.f, 0.f, 0.f, 0.f, 0.f};
  v8f acc[4] = {z, z, z, z};
  mm4<RD>(rpbf, RD, tempb + (size_t)m * RD * RD, RD, tm * 16, tg * 64, acc);
  float* D = tj + m * RD;
  int lane = threadIdx.x & 31;
  int row0 = tm * 16 + ((lane >> 4) << 3);
#pragma unroll
  for (int j = 0; j < 4; ++j) {
    int col = tg * 64 + j * 16 + (lane & 15);
#pragma unroll
    for (int v = 0; v < 8; ++v)
      D[(size_t)(row0 + v) * (MIXD * RD) + col] = acc[j][v];
  }
}

// gr2 = da1 @ hW1^T  (+ optional second-phase direct term)
__global__ void k_grad_r2(const u16* da1b, const u16* hW1b, float* gr2,
                          const float* r2, const float* r2b, const float* mf,
                          const float* denomPtr, int use2) {
  int wave = blockIdx.x * 8 + (threadIdx.x >> 5);  // 16 waves, grid 2
  int tm = wave >> 1, tg = wave & 1;
  v8f z = {0.f, 0.f, 0.f, 0.f, 0.f, 0.f, 0.f, 0.f};
  v8f acc[4] = {z, z, z, z};
  mm4<HIDD>(da1b, HIDD, hW1b, HIDD, tm * 16, tg * 64, acc);
  int lane = threadIdx.x & 31;
  int row0 = tm * 16 + ((lane >> 4) << 3);
#pragma unroll
  for (int j = 0; j < 4; ++j) {
    int col = tg * 64 + j * 16 + (lane & 15);
#pragma unroll
    for (int v = 0; v < 8; ++v) {
      int row = row0 + v;
      int idx = row * R2D + col;
      float g = acc[j][v];
      if (use2) {
        float denom = fmaxf(*denomPtr, 1.f);
        g += (2.0f / denom) * mf[row] * (r2[idx] - r2b[idx]);
      }
      gr2[idx] = g;
    }
  }
}

// ---------------- elementwise / reduction kernels ----------------
__global__ void k_ln_elu(const float* a1, const float* lng, const float* lnb,
                         u16* aebf, float* mu, float* rstd) {
  __shared__ float sb[256];
  int b = blockIdx.x, t = threadIdx.x;
  float x = a1[b * HIDD + t];
  float m = blk_sum(x, sb) * (1.f / 256.f);
  float d = x - m;
  float var = blk_sum(d * d, sb) * (1.f / 256.f);
  float rs = rsqrtf(var + 1e-5f);
  float aln = d * rs * lng[t] + lnb[t];
  float ev = (aln > 0.f) ? aln : (__expf(aln) - 1.f);
  aebf[b * HIDD + t] = f2bfbits(ev);
  if (t == 0) { mu[b] = m; rstd[b] = rs; }
}

__global__ void k_ln_bwd(const float* dae, const float* a1, const float* mu,
                         const float* rstd, const float* lng, const float* lnb,
                         u16* da1b) {
  __shared__ float sb[256];
  int b = blockIdx.x, t = threadIdx.x;
  int idx = b * HIDD + t;
  float xh = (a1[idx] - mu[b]) * rstd[b];
  float aln = xh * lng[t] + lnb[t];
  float dpre = dae[idx] * ((aln > 0.f) ? 1.f : __expf(aln));  // elu'
  float dxh = dpre * lng[t];
  float s1 = blk_sum(dxh, sb);
  float s2 = blk_sum(dxh * xh, sb);
  da1b[idx] = f2bfbits(rstd[b] * (dxh - s1 * (1.f / 256.f) - xh * (s2 * (1.f / 256.f))));
}

__global__ void k_pred(const float* w, const float* tj, float* pre) {
  int i = blockIdx.x * 256 + threadIdx.x;  // 128*256
  int b = i >> 8, c = i & 255;
  float s = 0.f;
#pragma unroll
  for (int m = 0; m < MIXD; ++m) s += w[b * MIXD + m] * tj[b * MIXD * RD + m * RD + c];
  pre[i] = s;
}

__global__ void k_drb(const float* pre, const float* r, float* drb,
                      float cnum, const float* denomPtr, const float* mf) {
  int i = blockIdx.x * 256 + threadIdx.x;
  int b = i >> 8;
  float denom = denomPtr ? fmaxf(*denomPtr, 1.f) : 128.f;
  float s = (cnum / denom) * (mf ? mf[b] : 1.f);
  float p = pre[i];
  drb[i] = (p > 0.f) ? s * (r[i] - p) : 0.f;
}

__global__ void k_gw(const float* drb, const float* tj, float* gw) {
  int i = blockIdx.x * 256 + threadIdx.x;  // 128*16
  int b = i >> 4, m = i & 15;
  const float* dp = drb + b * RD;
  const float* tp = tj + b * MIXD * RD + m * RD;
  float s = 0.f;
  for (int c = 0; c < RD; ++c) s += dp[c] * tp[c];
  gw[i] = s;
}

__global__ void k_dh3(const float* gw, const float* hW3, u16* da2b) {
  int i = blockIdx.x * 256 + threadIdx.x;  // 128*256
  int b = i >> 8, c = i & 255;
  float s = 0.f;
#pragma unroll
  for (int m = 0; m < MIXD; ++m) s += gw[b * MIXD + m] * hW3[c * MIXD + m];
  da2b[i] = f2bfbits(s);
}

__global__ void k_adam(const float* g, float* r2, u16* r2bf, float* m_, float* v_,
                       float* accs, const float* mf) {
  if (accs[4] != 0.f) return;
  int i = blockIdx.x * 256 + threadIdx.x;  // 128*128
  int b = i >> 7;
  float tn = accs[5] + 1.f;
  float gi = g[i];
  float m = 0.9f * m_[i] + 0.1f * gi;
  float v = 0.999f * v_[i] + 0.001f * gi * gi;
  float mh = m / (1.f - __powf(0.9f, tn));
  float vh = v / (1.f - __powf(0.999f, tn));
  float old = r2[i];
  float nw = old - 1e-2f * mh / (sqrtf(vh) + 1e-8f);
  m_[i] = m; v_[i] = v; r2[i] = nw; r2bf[i] = f2bfbits(nw);
  float wgt = mf ? mf[b] : 1.f;
  atomicAdd(&accs[2], wgt * (nw - old) * (nw - old));
  atomicAdd(&accs[3], wgt * old * old);
}

__global__ void k_conv(float* accs, int mode) {
  bool c = true;
  if (mode & 1) c = c && (sqrtf(accs[0]) / (sqrtf(accs[1]) + 1e-16f) < 1e-3f);
  if (mode & 2) c = c && (sqrtf(accs[2]) / (sqrtf(accs[3]) + 1e-16f) < 1e-3f);
  if (accs[4] == 0.f) {
    accs[5] += 1.f;
    if (c) accs[4] = 1.f;
  }
  accs[0] = accs[1] = accs[2] = accs[3] = 0.f;
}

__global__ void k_mask(const float* r, const float* prep, float* mf, float* accs) {
  __shared__ float sb[256];
  int b = blockIdx.x, t = threadIdx.x;
  float d = r[b * RD + t] - fmaxf(prep[b * RD + t], 0.f);
  float s = blk_sum(d * d, sb);
  if (t == 0) {
    float m = (s > 1.0f) ? 1.f : 0.f;  // THRES
    mf[b] = m;
    atomicAdd(&accs[6], m);
  }
}

__global__ void k_tloss(const float* r, const float* pre, float* accs) {
  __shared__ float sb[256];
  int b = blockIdx.x, t = threadIdx.x;
  float d = r[b * RD + t] - fmaxf(pre[b * RD + t], 0.f);
  float s = blk_sum(d * d, sb);
  if (t == 0) atomicAdd(&accs[7], s * (1.f / 128.f));
}

__global__ void k_t2loss(const float* r2, const float* r2b, const float* mf, float* accs) {
  __shared__ float sb[256];
  int b = blockIdx.x, t = threadIdx.x;
  float d = (t < R2D) ? (r2[b * R2D + t] - r2b[b * R2D + t]) : 0.f;
  float s = blk_sum(d * d, sb);
  if (t == 0) atomicAdd(&accs[8], mf[b] * s / fmaxf(accs[6], 1.f));
}

__global__ void k_scal(float* out, const float* accs) {
  out[0] = accs[7];
  out[1] = accs[8];
}

// ---------------- host orchestration ----------------
extern "C" void kernel_launch(void* const* d_in, const int* in_sizes, int n_in,
                              void* d_out, int out_size, void* d_ws, size_t ws_size,
                              hipStream_t stream) {
  (void)in_sizes; (void)n_in; (void)out_size; (void)ws_size;
  const float* X        = (const float*)d_in[0];
  const float* Wsd      = (const float*)d_in[1];
  const float* temporal = (const float*)d_in[2];
  const float* hW1 = (const float*)d_in[3];
  const float* hb1 = (const float*)d_in[4];
  const float* lng = (const float*)d_in[5];
  const float* lnb = (const float*)d_in[6];
  const float* hW2 = (const float*)d_in[7];
  const float* hb2 = (const float*)d_in[8];
  const float* hW3 = (const float*)d_in[9];
  const float* hb3 = (const float*)d_in[10];
  const float* t2W = (const float*)d_in[11];
  const float* t2b = (const float*)d_in[12];
  float* out = (float*)d_out;

  char* wp = (char*)d_ws;
  auto af = [&](size_t n) { float* p = (float*)wp; wp += ((n * 4 + 255) & ~(size_t)255); return p; };
  auto ab = [&](size_t n) { u16* p = (u16*)wp; wp += ((n * 2 + 255) & ~(size_t)255); return p; };

  float* r    = af(NB * RD);
  float* rf   = af(NB * RD);
  float* pre  = af(NB * RD);
  float* drb  = af(NB * RD);
  float* tj   = af((size_t)NB * MIXD * RD);
  float* a1   = af(NB * HIDD);
  float* dtmp = af(NB * HIDD);
  float* wbuf = af(NB * MIXD);
  float* gw   = af(NB * MIXD);
  float* gr2  = af(NB * R2D);
  float* r2   = af(NB * R2D);
  float* am   = af(NB * R2D);
  float* av   = af(NB * R2D);
  float* r2b  = af(NB * R2D);
  float* mu   = af(NB);
  float* rstd = af(NB);
  float* mf   = af(NB);
  float* accs = af(32);

  u16* r_bf   = ab(NB * RD);
  u16* rp_bf  = ab(NB * RD);
  u16* r2_bf  = ab(NB * R2D);
  u16* r2p_bf = ab(NB * R2D);
  u16* e_bf   = ab((size_t)NB * IND);
  u16* ae_bf  = ab(NB * HIDD);
  u16* a2_bf  = ab(NB * HIDD);
  u16* da2_bf = ab(NB * HIDD);
  u16* da1_bf = ab(NB * HIDD);

  u16* Wsd_bf  = ab((size_t)RD * IND);
  u16* WsdT_bf = ab((size_t)IND * RD);
  u16* temp_bf = ab((size_t)MIXD * RD * RD);
  u16* hW1_bf  = ab(R2D * HIDD);
  u16* hW1T_bf = ab(HIDD * R2D);
  u16* hW2_bf  = ab(HIDD * HIDD);
  u16* hW2T_bf = ab(HIDD * HIDD);
  u16* hW3T_bf = ab(MIXD * HIDD);
  u16* t2WT_bf = ab(R2D * R2D);

  auto zero = [&](float* p, int n) { kz<<<dim3((n + 255) / 256), dim3(256), 0, stream>>>(p, n); };
  auto zero16 = [&](u16* p, int n) { kz16<<<dim3((n + 255) / 256), dim3(256), 0, stream>>>(p, n); };
  auto cpy = [&](float* d, const float* s, int n) {
    kcopy<<<dim3((n + 255) / 256), dim3(256), 0, stream>>>(d, s, n);
  };
  auto cvt = [&](const float* s, u16* d, int n) {
    kcvt<<<dim3((n + 255) / 256), dim3(256), 0, stream>>>(s, d, n);
  };

  // ---- bf16 weight shadows (idempotent) ----
  cvt(Wsd, Wsd_bf, RD * IND);
  kcvtT<<<dim3(1024), dim3(256), 0, stream>>>(Wsd, WsdT_bf, RD, IND);
  cvt(temporal, temp_bf, MIXD * RD * RD);
  cvt(hW1, hW1_bf, R2D * HIDD);
  kcvtT<<<dim3(128), dim3(256), 0, stream>>>(hW1, hW1T_bf, R2D, HIDD);
  cvt(hW2, hW2_bf, HIDD * HIDD);
  kcvtT<<<dim3(256), dim3(256), 0, stream>>>(hW2, hW2T_bf, HIDD, HIDD);
  kcvtT<<<dim3(16), dim3(256), 0, stream>>>(hW3, hW3T_bf, HIDD, MIXD);
  kcvtT<<<dim3(64), dim3(256), 0, stream>>>(t2W, t2WT_bf, R2D, R2D);
  zero(accs, 32);

  auto hyper_fwd = [&](const u16* r2src) {
    k_gemm4<R2D, 4><<<4, 256, 0, stream>>>(r2src, R2D, hW1T_bf, R2D, hb1, a1, HIDD);
    k_ln_elu<<<128, 256, 0, stream>>>(a1, lng, lnb, ae_bf, mu, rstd);
    k_gemm4h<HIDD, 4><<<4, 256, 0, stream>>>(ae_bf, HIDD, hW2T_bf, HIDD, hb2, a2_bf, HIDD);
    k_h3<<<1, 256, 0, stream>>>(a2_bf, hW3T_bf, hb3, wbuf);
    k_pred<<<128, 256, 0, stream>>>(wbuf, tj, pre);
  };
  auto hyper_bwd = [&](const float* mfp, const float* denomPtr, int use2) {
    k_gw<<<8, 256, 0, stream>>>(drb, tj, gw);
    k_dh3<<<128, 256, 0, stream>>>(gw, hW3, da2_bf);
    k_gemm4<HIDD, 4><<<4, 256, 0, stream>>>(da2_bf, HIDD, hW2_bf, HIDD, nullptr, dtmp, HIDD);
    k_ln_bwd<<<128, 256, 0, stream>>>(dtmp, a1, mu, rstd, lng, lnb, da1_bf);
    k_grad_r2<<<2, 256, 0, stream>>>(da1_bf, hW1_bf, gr2, r2, r2b, mfp, denomPtr, use2);
  };

  // ---- _inf_first on X[:,0] ----
  zero(r, NB * RD);
  zero16(r_bf, NB * RD);
  zero(accs, 6);
  for (int it = 0; it < NITER; ++it) {
    k_err<<<16, 256, 0, stream>>>(X, NT * IND, r_bf, WsdT_bf, e_bf);
    k_update_r<<<4, 256, 0, stream>>>(e_bf, Wsd_bf, r, r_bf, pre, accs, 0);
    k_conv<<<1, 1, 0, stream>>>(accs, 1);
  }
  cpy(rf, r, NB * RD);
  cvt(r, rp_bf, NB * RD);
  zero(r2, NB * R2D);
  zero16(r2_bf, NB * R2D);

  auto run_inf = [&](int t) {
    zero(r, NB * RD);
    zero16(r_bf, NB * RD);
    zero(am, NB * R2D);
    zero(av, NB * R2D);
    zero(accs, 6);
    for (int it = 0; it < NITER; ++it) {
      hyper_fwd(r2_bf);
      k_err<<<16, 256, 0, stream>>>(X + t * IND, NT * IND, r_bf, WsdT_bf, e_bf);
      k_drb<<<128, 256, 0, stream>>>(pre, r, drb, -2.0f, nullptr, nullptr);
      k_update_r<<<4, 256, 0, stream>>>(e_bf, Wsd_bf, r, r_bf, pre, accs, 1);
      hyper_bwd(nullptr, nullptr, 0);
      k_adam<<<64, 256, 0, stream>>>(gr2, r2, r2_bf, am, av, accs, nullptr);
      k_conv<<<1, 1, 0, stream>>>(accs, 3);
    }
  };

  // ---- t = 1 ----
  k_tj<<<dim3(4, 16), 256, 0, stream>>>(rp_bf, temp_bf, tj);
  run_inf(1);

  // ---- t = 2..5 ----
  for (int t = 2; t < NT; ++t) {
    cvt(r, rp_bf, NB * RD);
    cvt(r2, r2p_bf, NB * R2D);
    k_tj<<<dim3(4, 16), 256, 0, stream>>>(rp_bf, temp_bf, tj);
    run_inf(t);

    // r2_loss mask from hyper(r2_p) against final r
    hyper_fwd(r2p_bf);
    zero(accs + 6, 1);
    k_mask<<<128, 256, 0, stream>>>(r, pre, mf, accs);

    // r2b = r2_p @ t2W + t2b (constant in second phase)
    k_gemm4<R2D, 2><<<2, 256, 0, stream>>>(r2p_bf, R2D, t2WT_bf, R2D, t2b, r2b, R2D);

    // _inf_second (in-place on r2; unmasked rows get zero grad)
    zero(am, NB * R2D);
    zero(av, NB * R2D);
    zero(accs, 6);
    for (int it = 0; it < NITER; ++it) {
      hyper_fwd(r2_bf);
      k_drb<<<128, 256, 0, stream>>>(pre, r, drb, -2.0f, accs + 6, mf);
      hyper_bwd(mf, accs + 6, 1);
      k_adam<<<64, 256, 0, stream>>>(gr2, r2, r2_bf, am, av, accs, mf);
      k_conv<<<1, 1, 0, stream>>>(accs, 2);
    }

    // losses for this timestep
    hyper_fwd(r2_bf);
    k_tloss<<<128, 256, 0, stream>>>(r, pre, accs);
    k_t2loss<<<128, 256, 0, stream>>>(r2, r2b, mf, accs);
  }

  // ---- outputs: [temp_loss, temp2_loss, r_first(128x256), r2(128x128)] ----
  k_scal<<<1, 1, 0, stream>>>(out, accs);
  cpy(out + 2, rf, NB * RD);
  cpy(out + 2 + NB * RD, r2, NB * R2D);
}
